// DepParser_41824391528575
// MI455X (gfx1250) — matrix operand order repaired and checked by reading, not compiled, not run on hardware
//
#include <hip/hip_runtime.h>

// ---------------------------------------------------------------------------
// DepParser pipeline for MI455X (gfx1250, wave32, WMMA)
//   N=512 seq, D=256 embed, H=512 hidden, out = 513x513 f32
// ---------------------------------------------------------------------------

typedef __attribute__((ext_vector_type(16))) __bf16 v16bf;
typedef __attribute__((ext_vector_type(8)))  __bf16 v8bf;
typedef __attribute__((ext_vector_type(8)))  float  v8f;

#define SEQ   512
#define DD    256          // D
#define TWO_D 512          // 2*D
#define HH    512          // H
#define G4    2048         // 4*H
#define ROWS  513          // SEQ + 1 (zeros row prepended)
#define MPAD  544          // ROWS padded to multiple of 32 (2x2 register blocking)

// Hardware transcendental tanh (gfx1250 V_TANH_F32). v_nop after the trans op
// per the ISA's multicycle co-execution hazard rule.
__device__ __forceinline__ float htanh(float x)
{
#if defined(__gfx1250__)
    float r;
    asm volatile("v_tanh_f32 %0, %1\n\tv_nop" : "=v"(r) : "v"(x));
    return r;
#else
    return tanhf(x);
#endif
}

// ---------------------------------------------------------------------------
// Prep kernels
// ---------------------------------------------------------------------------

__global__ void k_embed(const int* __restrict__ words, const int* __restrict__ pos,
                        const float* __restrict__ w_embed, const float* __restrict__ p_embed,
                        __bf16* __restrict__ x_bf)
{
    int t = blockIdx.x;          // 512 rows
    int d = threadIdx.x;         // 256 threads
    int w = words[t], p = pos[t];
    x_bf[(size_t)t * TWO_D + d]      = (__bf16)w_embed[(size_t)w * DD + d];
    x_bf[(size_t)t * TWO_D + DD + d] = (__bf16)p_embed[(size_t)p * DD + d];
}

__global__ void k_cvt_bf16(const float* __restrict__ src, __bf16* __restrict__ dst, int n)
{
    int i = blockIdx.x * blockDim.x + threadIdx.x;
    if (i < n) dst[i] = (__bf16)src[i];
}

// W' [1024][512] bf16 from fc1_w [512][1024]:
//   n <  512: W'[n][k] = fc1_w[n][k]         (Wa)
//   n >= 512: W'[n][k] = fc1_w[n-512][512+k] (Wb)
__global__ void k_build_wp(const float* __restrict__ fc1_w, __bf16* __restrict__ wp)
{
    int i = blockIdx.x * blockDim.x + threadIdx.x;
    if (i >= 1024 * HH) return;
    int n = i >> 9;
    int k = i & 511;
    float v = (n < HH) ? fc1_w[(size_t)n * (2 * HH) + k]
                       : fc1_w[(size_t)(n - HH) * (2 * HH) + HH + k];
    wp[i] = (__bf16)v;
}

__global__ void k_zero_bf16(__bf16* __restrict__ dst, int n)
{
    int i = blockIdx.x * blockDim.x + threadIdx.x;
    if (i < n) dst[i] = (__bf16)0.0f;
}

// ---------------------------------------------------------------------------
// WMMA fragment loads (wave32 layouts, ISA 7.12.2)
// ---------------------------------------------------------------------------
__device__ __forceinline__ v16bf load_afrag(const __bf16* p, int half)
{
    // A 16x32 bf16: elems 0..7 -> K = half*8 + e ; elems 8..15 -> K = 16 + half*8 + (e-8)
    v8bf lo = *(const v8bf*)(p + half * 8);
    v8bf hi = *(const v8bf*)(p + 16 + half * 8);
    return __builtin_shufflevector(lo, hi, 0,1,2,3,4,5,6,7,8,9,10,11,12,13,14,15);
}

__device__ __forceinline__ v16bf load_bfrag(const __bf16* p, int half)
{
    // B 32x16 bf16 (from row-major W): elems 0..7 -> K = half*8 + e ; elems 8..15 -> K = 16 + half*8 + (e-8)
    v8bf lo = *(const v8bf*)(p + half * 8);
    v8bf hi = *(const v8bf*)(p + 16 + half * 8);
    return __builtin_shufflevector(lo, hi, 0,1,2,3,4,5,6,7,8,9,10,11,12,13,14,15);
}

__device__ __forceinline__ v8f wmma_bf16(v16bf a, v16bf b, v8f c)
{
    return __builtin_amdgcn_wmma_f32_16x16x32_bf16(
        false, a, false, b, (short)0, c, false, false);
}

// ---------------------------------------------------------------------------
// WMMA GEMM:  C[M][N] = A[M][K] * B[N][K]^T (+ bias0[n] + bias1[n])
//   2x2 register blocking: each wave computes a 32x32 output tile
//   (2 A-frags x 2 B-frags -> 4 WMMAs per K-chunk). 8 waves/block along N.
//   Requires M % 32 == 0, N % 32 == 0, K % 32 == 0.
// ---------------------------------------------------------------------------
__global__ void k_gemm_bf16(const __bf16* __restrict__ A, const __bf16* __restrict__ B,
                            const float* __restrict__ bias0, const float* __restrict__ bias1,
                            float* __restrict__ C, int M, int N, int K)
{
    const int wave = threadIdx.x >> 5;
    const int lane = threadIdx.x & 31;
    const int half = lane >> 4;
    const int l16  = lane & 15;
    const int n0 = (blockIdx.x * 8 + wave) * 32;
    const int m0 = blockIdx.y * 32;

    const __bf16* arow0 = A + (size_t)(m0 + l16) * K;
    const __bf16* arow1 = A + (size_t)(m0 + 16 + l16) * K;
    const __bf16* brow0 = B + (size_t)(n0 + l16) * K;
    const __bf16* brow1 = B + (size_t)(n0 + 16 + l16) * K;

    v8f acc00 = {}, acc01 = {}, acc10 = {}, acc11 = {};
    for (int k0 = 0; k0 < K; k0 += 32) {
        v16bf a0 = load_afrag(arow0 + k0, half);
        v16bf a1 = load_afrag(arow1 + k0, half);
        v16bf b0 = load_bfrag(brow0 + k0, half);
        v16bf b1 = load_bfrag(brow1 + k0, half);
        acc00 = wmma_bf16(a0, b0, acc00);
        acc01 = wmma_bf16(a0, b1, acc01);
        acc10 = wmma_bf16(a1, b0, acc10);
        acc11 = wmma_bf16(a1, b1, acc11);
    }

    const int nA = n0 + l16;
    const int nB = nA + 16;
    float biasA = 0.0f, biasB = 0.0f;
    if (bias0) { biasA += bias0[nA]; biasB += bias0[nB]; }
    if (bias1) { biasA += bias1[nA]; biasB += bias1[nB]; }
    #pragma unroll
    for (int r = 0; r < 8; ++r) {
        int mA = m0 + r + half * 8;      // C/D layout: VGPR r -> M = r + 8*half
        int mB = mA + 16;
        C[(size_t)mA * N + nA] = acc00[r] + biasA;
        C[(size_t)mA * N + nB] = acc01[r] + biasB;
        C[(size_t)mB * N + nA] = acc10[r] + biasA;
        C[(size_t)mB * N + nB] = acc11[r] + biasB;
    }
}

// ---------------------------------------------------------------------------
// Sequential LSTM scan. Single 1024-thread workgroup (32 waves, one WGP).
// h, c, gates live in LDS; W_hh (4MB) streamed from L2 each step.
// ---------------------------------------------------------------------------
__global__ void __launch_bounds__(1024)
k_lstm_scan(const float* __restrict__ xg,    // [512][2048]
            const float* __restrict__ Whh,   // [2048][512]
            __bf16* __restrict__ out_bf)     // [544][512]
{
    __shared__ float h_s[HH];
    __shared__ float c_s[HH];
    __shared__ float gates[G4];

    const int tid = threadIdx.x;
    for (int j = tid; j < HH; j += blockDim.x) { h_s[j] = 0.0f; c_s[j] = 0.0f; }
    __syncthreads();

    for (int t = 0; t < SEQ; ++t) {
        for (int j = tid; j < G4; j += 1024) {
            const float* wrow = Whh + (size_t)j * HH;
            if (t + 1 < SEQ)
                __builtin_prefetch(xg + (size_t)(t + 1) * G4 + j, 0, 1);
            float s = xg[(size_t)t * G4 + j];
            #pragma unroll 8
            for (int k = 0; k < HH; ++k)
                s = fmaf(h_s[k], wrow[k], s);
            gates[j] = s;
        }
        __syncthreads();
        for (int j = tid; j < HH; j += 1024) {
            float ig = gates[j];
            float fg = gates[HH + j];
            float gg = gates[2 * HH + j];
            float og = gates[3 * HH + j];
            ig = 1.0f / (1.0f + __expf(-ig));
            fg = 1.0f / (1.0f + __expf(-fg));
            og = 1.0f / (1.0f + __expf(-og));
            gg = htanh(gg);
            float c = fg * c_s[j] + ig * gg;
            float h = og * htanh(c);
            c_s[j] = c;
            h_s[j] = h;
            out_bf[(size_t)(t + 1) * HH + j] = (__bf16)h;
        }
        __syncthreads();
    }
}

// ---------------------------------------------------------------------------
// Bilinear scoring: M[i,j] = sum_k tanh(A[i,k] + B[j,k] + fc1_b[k]) * fc2_w[k]
//                           + fc2_b.  AB = [544][1024], A=cols 0..511, B=cols 512..1023.
// 16x16 output tile per 256-thread block; K staged through LDS in chunks of 128.
// Inner loop: v_add / v_tanh_f32 / v_fma per k.
// ---------------------------------------------------------------------------
__global__ void k_score(const float* __restrict__ AB,
                        const float* __restrict__ fc1_b, const float* __restrict__ fc2_w,
                        const float* __restrict__ fc2_b,
                        float* __restrict__ Mout)
{
    __shared__ float As[16][128];
    __shared__ float Bs[16][128];
    __shared__ float bb[128];
    __shared__ float ww[128];

    const int tx = threadIdx.x & 15;     // j within tile
    const int ty = threadIdx.x >> 4;     // i within tile
    const int i0 = blockIdx.y * 16;
    const int j0 = blockIdx.x * 16;

    float acc = 0.0f;
    for (int k0 = 0; k0 < HH; k0 += 128) {
        for (int e = threadIdx.x; e < 16 * 128; e += 256) {
            int r = e >> 7, c = e & 127;
            int gi = i0 + r, gj = j0 + r;
            As[r][c] = (gi < ROWS) ? AB[(size_t)gi * 1024 + k0 + c] : 0.0f;
            Bs[r][c] = (gj < ROWS) ? AB[(size_t)gj * 1024 + HH + k0 + c] : 0.0f;
        }
        if (threadIdx.x < 128) {
            bb[threadIdx.x] = fc1_b[k0 + threadIdx.x];
            ww[threadIdx.x] = fc2_w[k0 + threadIdx.x];
        }
        __syncthreads();
        #pragma unroll 4
        for (int k = 0; k < 128; ++k)
            acc = fmaf(htanh(As[ty][k] + Bs[tx][k] + bb[k]), ww[k], acc);
        __syncthreads();
    }

    const int i = i0 + ty, j = j0 + tx;
    if (i < ROWS && j < ROWS)
        Mout[(size_t)i * ROWS + j] = acc + fc2_b[0];
}

// ---------------------------------------------------------------------------
// Launch
// ---------------------------------------------------------------------------
extern "C" void kernel_launch(void* const* d_in, const int* in_sizes, int n_in,
                              void* d_out, int out_size, void* d_ws, size_t ws_size,
                              hipStream_t stream)
{
    const int*   words   = (const int*)d_in[0];
    const int*   pos     = (const int*)d_in[1];
    const float* w_embed = (const float*)d_in[2];
    const float* p_embed = (const float*)d_in[3];
    const float* W_ih    = (const float*)d_in[4];
    const float* W_hh    = (const float*)d_in[5];
    const float* b_ih    = (const float*)d_in[6];
    const float* b_hh    = (const float*)d_in[7];
    const float* fc1_w   = (const float*)d_in[8];
    const float* fc1_b   = (const float*)d_in[9];
    const float* fc2_w   = (const float*)d_in[10];
    const float* fc2_b   = (const float*)d_in[11];
    float* Mout = (float*)d_out;

    char* ws = (char*)d_ws;
    __bf16* x_bf   = (__bf16*)(ws + 0);                       // 512*512*2   = 524288
    __bf16* wih_bf = (__bf16*)(ws + 524288);                  // 2048*512*2  = 2097152
    __bf16* wp_bf  = (__bf16*)(ws + 2621440);                 // 1024*512*2  = 1048576
    float*  xg     = (float*) (ws + 3670016);                 // 512*2048*4  = 4194304
    __bf16* out_bf = (__bf16*)(ws + 7864320);                 // 544*512*2   = 557056
    float*  AB     = (float*) (ws + 8421376);                 // 544*1024*4  = 2228224

    // --- prep ---
    k_embed<<<dim3(SEQ), dim3(DD), 0, stream>>>(words, pos, w_embed, p_embed, x_bf);
    {
        int n = G4 * TWO_D;
        k_cvt_bf16<<<dim3((n + 255) / 256), dim3(256), 0, stream>>>(W_ih, wih_bf, n);
    }
    {
        int n = 1024 * HH;
        k_build_wp<<<dim3((n + 255) / 256), dim3(256), 0, stream>>>(fc1_w, wp_bf);
    }
    {
        int n = MPAD * HH;
        k_zero_bf16<<<dim3((n + 255) / 256), dim3(256), 0, stream>>>(out_bf, n);
    }

    // --- xg = x @ W_ih^T + b_ih + b_hh   [512, 2048], K=512 (WMMA bf16) ---
    k_gemm_bf16<<<dim3(G4 / 32 / 8, SEQ / 32), dim3(256), 0, stream>>>(
        x_bf, wih_bf, b_ih, b_hh, xg, SEQ, G4, TWO_D);

    // --- sequential LSTM scan (fills out_bf rows 1..512) ---
    k_lstm_scan<<<dim3(1), dim3(1024), 0, stream>>>(xg, W_hh, out_bf);

    // --- AB = out_pad @ W'^T   [544, 1024], K=512 (WMMA bf16) ---
    k_gemm_bf16<<<dim3(1024 / 32 / 8, MPAD / 32), dim3(256), 0, stream>>>(
        out_bf, wp_bf, nullptr, nullptr, AB, MPAD, 1024, HH);

    // --- bilinear tanh scoring -> M [513, 513] ---
    k_score<<<dim3((ROWS + 15) / 16, (ROWS + 15) / 16), dim3(256), 0, stream>>>(
        AB, fc1_b, fc2_w, fc2_b, Mout);
}
